// HANLayer_79834852098259
// MI455X (gfx1250) — matrix operand order repaired and checked by reading, not compiled, not run on hardware
//
#include <hip/hip_runtime.h>
#include <hip/hip_bf16.h>

// Problem constants (match reference)
#define NN 20000
#define EE 640000
#define IIN 256
#define HH 8
#define DD 16
#define MM 3
#define HD 128   // H*D
#define HID 128

typedef __attribute__((ext_vector_type(16))) __bf16 v16bf;
typedef __attribute__((ext_vector_type(8)))  float  v8f;
typedef unsigned u32x4 __attribute__((ext_vector_type(4)));
typedef int      i32x4 __attribute__((ext_vector_type(4)));
typedef int      i32x8 __attribute__((ext_vector_type(8)));

#if defined(__HIP_DEVICE_COMPILE__) && __has_builtin(__builtin_amdgcn_tensor_load_to_lds)
#define HAS_TDM 1
#else
#define HAS_TDM 0
#endif

// ---------------- helpers ----------------
__device__ __forceinline__ unsigned f2key(float f) {
    unsigned b = __float_as_uint(f);
    return (b & 0x80000000u) ? ~b : (b | 0x80000000u);
}
__device__ __forceinline__ float key2f(unsigned k) {
    if (k == 0u) return 0.0f;  // empty segment -> 0 (matches isfinite() fixup)
    unsigned b = (k & 0x80000000u) ? (k & 0x7FFFFFFFu) : ~k;
    return __uint_as_float(b);
}

#if HAS_TDM
// Tensor DMA Descriptor builders (CDNA5 ISA).
// Group 0: count=1 | lds_addr[63:32] | global_addr[120:64] | type=2 [127:126]
__device__ __forceinline__ u32x4 tdm_g0(unsigned lds_off, unsigned long long ga) {
    u32x4 g;
    g.x = 1u;                                   // count=1, is_restore=0, gather off
    g.y = lds_off;                              // LDS byte address of tile start
    g.z = (unsigned)ga;                         // global_addr[31:0]
    g.w = (unsigned)((ga >> 32) & 0x01FFFFFFu)  // global_addr[56:32]
          | 0x80000000u;                        // type = 2 ("image")
    return g;
}
// Group 1 for a plain 2D tile, data_size = 2 bytes (bf16).
// tensor dims == tile dims (tile exactly covers the described tensor -> no OOB).
__device__ __forceinline__ i32x8 tdm_g1_2d(unsigned tdim0, unsigned tdim1,
                                           unsigned tile0, unsigned tile1,
                                           unsigned stride0) {
    i32x8 g;
    g[0] = 0x00010000;                                    // data_size=1 (2B); no flags
    g[1] = (int)((tdim0 & 0xFFFFu) << 16);                // tensor_dim0[15:0]
    g[2] = (int)((tdim0 >> 16) | ((tdim1 & 0xFFFFu) << 16)); // tdim0 hi | tdim1 lo
    g[3] = (int)((tdim1 >> 16) | (tile0 << 16));          // tdim1 hi | tile_dim0
    g[4] = (int)(tile1 & 0xFFFFu);                        // tile_dim1 ; tile_dim2=0
    g[5] = (int)stride0;                                  // tensor_dim0_stride[31:0]
    g[6] = 0;                                             // stride0 hi | stride1 lo
    g[7] = 0;
    return g;
}
__device__ __forceinline__ void tdm_load(u32x4 g0, i32x8 g1) {
    i32x4 z4 = {0, 0, 0, 0};
#if __clang_major__ >= 23
    i32x8 z8 = {0, 0, 0, 0, 0, 0, 0, 0};
    __builtin_amdgcn_tensor_load_to_lds(g0, g1, z4, z4, z8, 0);
#else
    __builtin_amdgcn_tensor_load_to_lds(g0, g1, z4, z4, 0);
#endif
}
#endif // HAS_TDM

// ---------------- fp32 -> bf16 convert (h) ----------------
__global__ void conv_bf16_kernel(const float* __restrict__ in, __bf16* __restrict__ out, int n) {
    int i = blockIdx.x * blockDim.x + threadIdx.x;
    if (i < n) out[i] = (__bf16)in[i];
}

// ---------------- pack W: fp32 [M][K=256][N=128] -> bf16 transposed [M][N=128][K=256] ----
__global__ void packW_kernel(const float* __restrict__ W, __bf16* __restrict__ Wt) {
    int idx = blockIdx.x * blockDim.x + threadIdx.x;          // m*128*256 + n*256 + k
    if (idx >= MM * HD * IIN) return;
    int k = idx & 255;
    int n = (idx >> 8) & 127;
    int m = idx >> 15;
    Wt[idx] = (__bf16)W[((size_t)m * IIN + k) * HD + n];
}

// ---------------- WMMA GEMM: ft[n, m, :] = h[n,:] @ W[m,:,:] ----------------
// grid = (N/32, M), block = 256 (8 waves). Each wave owns 16 output cols and
// two 16-row node sub-tiles. Whole A tile (32x256, 16KB) and whole W[m]
// (128x256, 64KB) staged into LDS by two Tensor Data Mover ops -> 80KB LDS,
// exactly 4 workgroups per 320KB WGP. Inner loop is pure ds_load_b128 + wmma.
__global__ void __launch_bounds__(256)
gat_gemm_wmma(const __bf16* __restrict__ hb,   // [N][256] bf16
              const __bf16* __restrict__ Wt,   // [M][128][256] bf16 (col-major W)
              float* __restrict__ z)           // [N][M][128] f32
{
    const int node0 = blockIdx.x * 32;
    const int m     = blockIdx.y;

    __shared__ __bf16 Alds[32][256];      // node tile (row-major, TDM-compact) 16KB
    __shared__ __bf16 Blds[128][256];     // full W[m], [n][k] (row-major)      64KB

    const int lane = threadIdx.x & 31;
    const int wv   = threadIdx.x >> 5;    // 0..7 -> output col group
    const int half = lane >> 4;           // 0/1
    const int l16  = lane & 15;

#if HAS_TDM
    if (threadIdx.x < 32) {
        // A tile: 32 rows x 256 cols of bf16, row stride 256 elements
        unsigned long long ga = (unsigned long long)(const void*)(hb + (size_t)node0 * IIN);
        tdm_load(tdm_g0((unsigned)(uintptr_t)&Alds[0][0], ga),
                 tdm_g1_2d(/*tdim0=*/IIN, /*tdim1=*/32, /*tile0=*/IIN, /*tile1=*/32,
                           /*stride0=*/IIN));
        // Full B: 128 rows (n) x 256 cols (k) from Wt[m]
        unsigned long long gb = (unsigned long long)(const void*)(Wt + (size_t)m * HD * IIN);
        tdm_load(tdm_g0((unsigned)(uintptr_t)&Blds[0][0], gb),
                 tdm_g1_2d(/*tdim0=*/IIN, /*tdim1=*/HD, /*tile0=*/IIN, /*tile1=*/HD,
                           /*stride0=*/IIN));
        __builtin_amdgcn_s_wait_tensorcnt(0);
    }
#else
    for (int e = threadIdx.x; e < 32 * 256; e += 256) {
        int r = e >> 8, k = e & 255;
        Alds[r][k] = hb[(size_t)(node0 + r) * IIN + k];
    }
    for (int e = threadIdx.x; e < 128 * 256; e += 256) {
        int n = e >> 8, k = e & 255;
        Blds[n][k] = Wt[(size_t)m * HD * IIN + (size_t)n * IIN + k];
    }
#endif
    __syncthreads();                       // LDS tiles visible to all 8 waves

    v8f acc0 = {};                         // nodes node0 + 0..15
    v8f acc1 = {};                         // nodes node0 + 16..31

#pragma unroll
    for (int kk = 0; kk < 8; ++kk) {
        const int k0 = kk * 32;
        // fragments per ISA 7.12.2 (wave32, 16-bit A 16x32 / B 32x16)
        v16bf b;
#pragma unroll
        for (int v = 0; v < 8; ++v) {
            int kbb = ((v < 4) ? 2 * v : 2 * v + 8) + 8 * half;  // B K index
            b[2 * v]     = Blds[16 * wv + l16][k0 + kbb];
            b[2 * v + 1] = Blds[16 * wv + l16][k0 + kbb + 1];
        }
        v16bf a0, a1;
#pragma unroll
        for (int v = 0; v < 8; ++v) {
            int kb = ((v < 4) ? 2 * v : 2 * v + 8) + 8 * half;   // A K index
            a0[2 * v]     = Alds[l16][k0 + kb];
            a0[2 * v + 1] = Alds[l16][k0 + kb + 1];
            a1[2 * v]     = Alds[16 + l16][k0 + kb];
            a1[2 * v + 1] = Alds[16 + l16][k0 + kb + 1];
        }
        acc0 = __builtin_amdgcn_wmma_f32_16x16x32_bf16(
            false, a0, false, b, (short)0, acc0, false, false);
        acc1 = __builtin_amdgcn_wmma_f32_16x16x32_bf16(
            false, a1, false, b, (short)0, acc1, false, false);
    }

    // D layout: VGPR r -> row r + 8*half, lane%16 -> col
#pragma unroll
    for (int r = 0; r < 8; ++r) {
        int row0 = node0 + r + 8 * half;
        z[((size_t)row0 * MM + m) * HD + 16 * wv + l16] = acc0[r];
        int row1 = node0 + 16 + r + 8 * half;
        z[((size_t)row1 * MM + m) * HD + 16 * wv + l16] = acc1[r];
    }
}

// ---------------- el/er: per (m, n, h) dot over D ----------------
__global__ void ler_kernel(const float* __restrict__ z,
                           const float* __restrict__ attn_l,  // [M][H][D]
                           const float* __restrict__ attn_r,
                           float* __restrict__ el,            // [M][N][H]
                           float* __restrict__ er)
{
    int idx = blockIdx.x * blockDim.x + threadIdx.x;
    if (idx >= MM * NN * HH) return;
    int h = idx & 7;
    int n = (idx >> 3) % NN;
    int m = idx / (NN * HH);
    const float* ft  = z + ((size_t)n * MM + m) * HD + h * DD;
    const float* alp = attn_l + (m * HH + h) * DD;
    const float* arp = attn_r + (m * HH + h) * DD;
    float sl = 0.f, sr = 0.f;
#pragma unroll
    for (int d = 0; d < DD; ++d) { sl += ft[d] * alp[d]; sr += ft[d] * arp[d]; }
    el[((size_t)m * NN + n) * HH + h] = sl;
    er[((size_t)m * NN + n) * HH + h] = sr;
}

// ---------------- edge pass 1: scores + segment max (keyed atomicMax) ----------------
__global__ void edge_pass1(const int* __restrict__ src, const int* __restrict__ dst,
                           const float* __restrict__ edge_w,
                           const float* __restrict__ el, const float* __restrict__ er,
                           float* __restrict__ est,          // [M*E][H]
                           unsigned* __restrict__ mx)        // [M][N][H] keyed
{
    int idx = blockIdx.x * blockDim.x + threadIdx.x;
    if (idx >= MM * EE) return;
    int m = idx / EE;
    int s = src[idx], d = dst[idx];
    float ew = edge_w[idx];
    const float* elp = el + ((size_t)m * NN + s) * HH;
    const float* erp = er + ((size_t)m * NN + d) * HH;
    float* ep = est + (size_t)idx * HH;
    unsigned* mxp = mx + ((size_t)m * NN + d) * HH;
#pragma unroll
    for (int h = 0; h < HH; ++h) {
        float v = (elp[h] + erp[h]) * ew;
        v = v > 0.f ? v : 0.2f * v;       // LeakyReLU(0.2)
        ep[h] = v;
        atomicMax(&mxp[h], f2key(v));
    }
}

// ---------------- edge pass 2: exp + denom ----------------
__global__ void edge_pass2(const int* __restrict__ dst,
                           float* __restrict__ est,
                           const unsigned* __restrict__ mx,
                           float* __restrict__ denom)        // [M][N][H]
{
    int idx = blockIdx.x * blockDim.x + threadIdx.x;
    if (idx >= MM * EE) return;
    int m = idx / EE;
    int d = dst[idx];
    float* ep = est + (size_t)idx * HH;
    const unsigned* mxp = mx + ((size_t)m * NN + d) * HH;
    float* dnp = denom + ((size_t)m * NN + d) * HH;
#pragma unroll
    for (int h = 0; h < HH; ++h) {
        float ex = __expf(ep[h] - key2f(mxp[h]));
        ep[h] = ex;
        atomicAdd(&dnp[h], ex);
    }
}

// ---------------- edge pass 3: weighted scatter-add (1 edge per 128-thread block) ----
__global__ void __launch_bounds__(128)
edge_pass3(const int* __restrict__ src, const int* __restrict__ dst,
           const float* __restrict__ est, const float* __restrict__ denom,
           const float* __restrict__ z,
           float* __restrict__ acc)                          // [M][N][128]
{
    int b = blockIdx.x;                   // m*E + e
    int m = b / EE;
    int s = src[b], d = dst[b];
    int t = threadIdx.x;                  // 0..127 -> (h = t>>4, d = t&15)
    int h = t >> 4;
    float a  = est[(size_t)b * HH + h] / denom[((size_t)m * NN + d) * HH + h];
    float ft = z[((size_t)s * MM + m) * HD + t];
    atomicAdd(&acc[((size_t)m * NN + d) * HD + t], ft * a);
}

// ---------------- epilogue: ELU(acc + bias) -> overwrite z ----------------
__global__ void elu_kernel(const float* __restrict__ acc, const float* __restrict__ bias,
                           float* __restrict__ z)
{
    int idx = blockIdx.x * blockDim.x + threadIdx.x;
    if (idx >= NN * MM * HD) return;
    int f = idx % HD;
    int m = (idx / HD) % MM;
    int n = idx / (MM * HD);
    float v = acc[((size_t)m * NN + n) * HD + f] + bias[m * HD + f];
    v = v > 0.f ? v : expm1f(v);
    z[(size_t)n * (MM * HD) + m * HD + f] = v;
}

// ---------------- semantic attention scores: wsum[m] = mean_n w2 . tanh(z W1 + b1) ----
__global__ void sem_kernel(const float* __restrict__ z, const float* __restrict__ W1,
                           const float* __restrict__ b1, const float* __restrict__ w2,
                           float* __restrict__ wsum)
{
    int idx = blockIdx.x * blockDim.x + threadIdx.x;
    if (idx >= NN * MM) return;
    int n = idx / MM, m = idx % MM;
    const float* zr = z + (size_t)n * (MM * HD) + m * HD;
    float a = 0.f;
    for (int j = 0; j < HID; ++j) {
        float s = b1[j];
        for (int f = 0; f < HD; ++f) s += zr[f] * W1[f * HID + j];
        a += tanhf(s) * w2[j];
    }
    atomicAdd(&wsum[m], a * (1.0f / NN));
}

// ---------------- softmax over M=3 ----------------
__global__ void beta_kernel(const float* __restrict__ wsum, float* __restrict__ beta) {
    float w0 = wsum[0], w1 = wsum[1], w2 = wsum[2];
    float mx = fmaxf(w0, fmaxf(w1, w2));
    float e0 = __expf(w0 - mx), e1 = __expf(w1 - mx), e2 = __expf(w2 - mx);
    float s = e0 + e1 + e2;
    beta[0] = e0 / s; beta[1] = e1 / s; beta[2] = e2 / s;
}

// ---------------- final combine ----------------
__global__ void combine_kernel(const float* __restrict__ z, const float* __restrict__ beta,
                               float* __restrict__ out)
{
    int idx = blockIdx.x * blockDim.x + threadIdx.x;
    if (idx >= NN * HD) return;
    int n = idx >> 7, f = idx & 127;
    const float* zr = z + (size_t)n * (MM * HD);
    out[idx] = zr[f] * beta[0] + zr[HD + f] * beta[1] + zr[2 * HD + f] * beta[2];
}

// =================================================================
extern "C" void kernel_launch(void* const* d_in, const int* in_sizes, int n_in,
                              void* d_out, int out_size, void* d_ws, size_t ws_size,
                              hipStream_t stream) {
    const float* h      = (const float*)d_in[0];
    const float* fc_W   = (const float*)d_in[1];
    const float* attn_l = (const float*)d_in[2];
    const float* attn_r = (const float*)d_in[3];
    const float* bias   = (const float*)d_in[4];
    const float* edge_w = (const float*)d_in[5];
    const float* sem_W1 = (const float*)d_in[6];
    const float* sem_b1 = (const float*)d_in[7];
    const float* sem_w2 = (const float*)d_in[8];
    const int*   src    = (const int*)d_in[9];
    const int*   dst    = (const int*)d_in[10];
    float* out = (float*)d_out;

    // workspace carve-up (256B aligned)
    char* ws = (char*)d_ws;
    size_t off = 0;
    auto carve = [&](size_t bytes) -> char* {
        char* p = ws + off;
        off = (off + bytes + 255) & ~(size_t)255;
        return p;
    };
    __bf16*   hb    = (__bf16*)  carve((size_t)NN * IIN * 2);
    __bf16*   Wt    = (__bf16*)  carve((size_t)MM * HD * IIN * 2);  // transposed bf16 W
    float*    z     = (float*)   carve((size_t)NN * MM * HD * 4);
    float*    el    = (float*)   carve((size_t)MM * NN * HH * 4);
    float*    er    = (float*)   carve((size_t)MM * NN * HH * 4);
    float*    est   = (float*)   carve((size_t)MM * EE * HH * 4);
    // zero-initialized region (single memset): mx, denom, acc, wsum
    char*     zbase = ws + off;
    unsigned* mx    = (unsigned*)carve((size_t)MM * NN * HH * 4);
    float*    denom = (float*)   carve((size_t)MM * NN * HH * 4);
    float*    accb  = (float*)   carve((size_t)MM * NN * HD * 4);
    float*    wsum  = (float*)   carve(16);
    size_t    zbytes = (size_t)((ws + off) - zbase);
    float*    beta  = (float*)   carve(16);

    hipMemsetAsync(zbase, 0, zbytes, stream);

    // 1) convert h to bf16; pack W to transposed bf16
    {
        int n1 = NN * IIN;
        conv_bf16_kernel<<<(n1 + 255) / 256, 256, 0, stream>>>(h, hb, n1);
        int n2 = MM * HD * IIN;
        packW_kernel<<<(n2 + 255) / 256, 256, 0, stream>>>(fc_W, Wt);
    }
    // 2) WMMA GEMM (TDM-staged) -> ft (z)
    {
        dim3 grid(NN / 32, MM);
        gat_gemm_wmma<<<grid, 256, 0, stream>>>(hb, Wt, z);
    }
    // 3) el / er
    {
        int n = MM * NN * HH;
        ler_kernel<<<(n + 255) / 256, 256, 0, stream>>>(z, attn_l, attn_r, el, er);
    }
    // 4) edge softmax passes
    {
        int n = MM * EE;
        edge_pass1<<<(n + 255) / 256, 256, 0, stream>>>(src, dst, edge_w, el, er, est, mx);
        edge_pass2<<<(n + 255) / 256, 256, 0, stream>>>(dst, est, mx, denom);
        edge_pass3<<<n, 128, 0, stream>>>(src, dst, est, denom, z, accb);
    }
    // 5) ELU epilogue -> z holds per-metapath embeddings
    {
        int n = NN * MM * HD;
        elu_kernel<<<(n + 255) / 256, 256, 0, stream>>>(accb, bias, z);
    }
    // 6) semantic attention
    {
        int n = NN * MM;
        sem_kernel<<<(n + 127) / 128, 128, 0, stream>>>(z, sem_W1, sem_b1, sem_w2, wsum);
        beta_kernel<<<1, 1, 0, stream>>>(wsum, beta);
    }
    // 7) combine
    {
        int n = NN * HD;
        combine_kernel<<<(n + 255) / 256, 256, 0, stream>>>(z, beta, out);
    }
}
